// RecurrentGCN_67224828117589
// MI455X (gfx1250) — matrix operand order repaired and checked by reading, not compiled, not run on hardware
//
#include <hip/hip_runtime.h>
#include <hip/hip_bf16.h>

typedef __attribute__((ext_vector_type(16))) _Float16 v16h;
typedef __attribute__((ext_vector_type(8)))  _Float16 v8h;
typedef __attribute__((ext_vector_type(8)))  float    v8f;

#define HID   128
#define FIN   16
#define KPAD  160      // 32 (x padded) + 128 (h)
#define GCOLS 512      // 4 gates * 128
#define TM    32       // nodes per block
#define RT    2        // row tiles of 16
#define BLOCK 256

// workspace layout (bytes)
#define WS_WTG 0                          // f16 [512 cols][160 k]   (col-major weights, K contiguous)
#define WS_WT1 (WS_WTG + GCOLS*KPAD*2)    // f16 [64 cols][128 k]
#define WS_WT2 (WS_WT1 + 64*128*2)        // f16 [64 cols][64 k]
#define WS_BG  (WS_WT2 + 64*64*2)         // f32 [512]  (conv_b + b_gate fused)
#define PREP_T (GCOLS*KPAD + 64*128 + 64*64 + GCOLS)

// ---------------------------------------------------------------- prep: f32 -> f16 repack
__global__ void gclstm_prep(const float* __restrict__ Wi, const float* __restrict__ Wf,
                            const float* __restrict__ Wc, const float* __restrict__ Wo,
                            const float* __restrict__ Ci, const float* __restrict__ Cf,
                            const float* __restrict__ Cc, const float* __restrict__ Co,
                            const float* __restrict__ cbi, const float* __restrict__ cbf,
                            const float* __restrict__ cbc, const float* __restrict__ cbo,
                            const float* __restrict__ bi,  const float* __restrict__ bf,
                            const float* __restrict__ bc,  const float* __restrict__ bo,
                            const float* __restrict__ w1,  const float* __restrict__ w2,
                            char* __restrict__ ws) {
    int idx = blockIdx.x * blockDim.x + threadIdx.x;
    _Float16* WTG = (_Float16*)(ws + WS_WTG);
    _Float16* WT1 = (_Float16*)(ws + WS_WT1);
    _Float16* WT2 = (_Float16*)(ws + WS_WT2);
    float*    BG  = (float*)(ws + WS_BG);
    if (idx < GCOLS * KPAD) {
        int col = idx / KPAD, k = idx % KPAD;
        int g = col >> 7, j = col & 127;
        const float* W = (g == 0) ? Wi : (g == 1) ? Wf : (g == 2) ? Wc : Wo;
        const float* C = (g == 0) ? Ci : (g == 1) ? Cf : (g == 2) ? Cc : Co;
        float v = 0.0f;
        if (k < FIN)      v = W[k * HID + j];
        else if (k >= 32) v = C[(k - 32) * HID + j];
        WTG[idx] = (_Float16)v;
    } else if (idx < GCOLS * KPAD + 64 * 128) {
        int li = idx - GCOLS * KPAD;
        int col = li >> 7, k = li & 127;
        WT1[li] = (_Float16)w1[k * 64 + col];
    } else if (idx < GCOLS * KPAD + 64 * 128 + 64 * 64) {
        int li = idx - GCOLS * KPAD - 64 * 128;
        int col = li >> 6, k = li & 63;
        WT2[li] = (_Float16)w2[k * 64 + col];
    } else if (idx < PREP_T) {
        int li = idx - GCOLS * KPAD - 64 * 128 - 64 * 64;
        int g = li >> 7, j = li & 127;
        const float* cb = (g == 0) ? cbi : (g == 1) ? cbf : (g == 2) ? cbc : cbo;
        const float* b  = (g == 0) ? bi  : (g == 1) ? bf  : (g == 2) ? bc  : bo;
        BG[li] = cb[j] + b[j];
    }
}

// A-fragment in contiguous K order (matches B): lane holds K = 16*hi..16*hi+15,
// hi = lane>>4.
__device__ inline v16h load_a_frag(const _Float16* __restrict__ rowk0, int hi) {
    v8h lo = *(const v8h*)(rowk0 + 16 * hi);
    v8h hv = *(const v8h*)(rowk0 + 16 * hi + 8);
    v16h a;
#pragma unroll
    for (int i = 0; i < 8; ++i) { a[i] = lo[i]; a[i + 8] = hv[i]; }
    return a;
}

__device__ inline float sigf(float v) { return 1.0f / (1.0f + __expf(-v)); }

// ---------------------------------------------------------------- fused GCLSTM + MLP
__global__ __launch_bounds__(BLOCK) void gclstm_main(
    const float* __restrict__ x, const float* __restrict__ h, const float* __restrict__ c,
    const float* __restrict__ wci, const float* __restrict__ wcf, const float* __restrict__ wco,
    const float* __restrict__ mb1, const float* __restrict__ mb2,
    const float* __restrict__ w3,  const float* __restrict__ b3,
    const char* __restrict__ ws,
    float* __restrict__ out, float* __restrict__ hout, float* __restrict__ cout, int N) {

    __shared__ _Float16 ALDS[TM * KPAD];  // [32][160] f16 : x | pad | h
    __shared__ _Float16 HLDS[TM * HID];   // relu(h_new) f16
    __shared__ _Float16 M2[TM * 64];
    __shared__ _Float16 M3[TM * 64];

    const _Float16* WTG = (const _Float16*)(ws + WS_WTG);
    const _Float16* WT1 = (const _Float16*)(ws + WS_WT1);
    const _Float16* WT2 = (const _Float16*)(ws + WS_WT2);
    const float*    BG  = (const float*)(ws + WS_BG);

    const int tid = threadIdx.x;
    const int n0  = blockIdx.x * TM;

    // ---- stage [x | 0 | h] tile as f16 (uniform trip count -> EXEC stays full)
#pragma unroll
    for (int i = tid; i < TM * KPAD; i += BLOCK) {
        int m = i / KPAD, k = i % KPAD;
        int n = n0 + m; if (n >= N) n = N - 1;
        float v = (k < FIN) ? x[n * FIN + k] : ((k < 32) ? 0.0f : h[n * HID + (k - 32)]);
        ALDS[i] = (_Float16)v;
    }
    __syncthreads();

    const int wv = tid >> 5, lane = tid & 31;
    const int ln = lane & 15, hi = lane >> 4;
    const int j = wv * 16 + ln;            // this wave owns hidden cols [16*wv, 16*wv+16)

    v8f acc[RT][4];
#pragma unroll
    for (int r = 0; r < RT; ++r)
#pragma unroll
        for (int g = 0; g < 4; ++g) acc[r][g] = (v8f)(0.0f);

    // ---- all four gate GEMMs, K = 160 (5 tiles of 32)
#pragma unroll
    for (int kt = 0; kt < 5; ++kt) {
        const int k0 = kt * 32;
        v16h a[RT];
#pragma unroll
        for (int r = 0; r < RT; ++r)
            a[r] = load_a_frag(&ALDS[(r * 16 + ln) * KPAD + k0], hi);
#pragma unroll
        for (int g = 0; g < 4; ++g) {
            v16h bf = *(const v16h*)&WTG[(g * 128 + j) * KPAD + k0 + 16 * hi];
#pragma unroll
            for (int r = 0; r < RT; ++r)
                acc[r][g] = __builtin_amdgcn_wmma_f32_16x16x32_f16(
                    false, a[r], false, bf, (short)0, acc[r][g], false, false);
        }
    }

    // ---- LSTM pointwise, fully in-register (gates share (node, j) per lane)
    const float wciv = wci[j], wcfv = wcf[j], wcov = wco[j];
    const float bI = BG[j], bF = BG[128 + j], bC = BG[256 + j], bO = BG[384 + j];
#pragma unroll
    for (int r = 0; r < RT; ++r) {
#pragma unroll
        for (int v = 0; v < 8; ++v) {
            const int mblk = r * 16 + hi * 8 + v;   // C/D layout: M = vgpr + 8*hi
            const int n = n0 + mblk;
            const int nc = (n < N) ? n : (N - 1);
            const float cv = c[nc * HID + j];
            const float iv = sigf(acc[r][0][v] + bI + wciv * cv);
            const float fv = sigf(acc[r][1][v] + bF + wcfv * cv);
            const float tv = tanhf(acc[r][2][v] + bC);
            const float cn = fv * cv + iv * tv;
            const float ov = sigf(acc[r][3][v] + bO + wcov * cn);
            const float hn = ov * tanhf(cn);
            if (n < N) { cout[n * HID + j] = cn; hout[n * HID + j] = hn; }
            HLDS[mblk * HID + j] = (_Float16)fmaxf(hn, 0.0f);
        }
    }
    __syncthreads();

    // ---- MLP layer 1: [32,128] @ [128,64]  (8 tasks = 2 row-tiles x 4 col-tiles)
    const int r1  = wv >> 2;
    const int cj  = (wv & 3) * 16 + ln;    // output col 0..63
    v8f m1 = (v8f)(0.0f);
#pragma unroll
    for (int kt = 0; kt < 4; ++kt) {
        const int k0 = kt * 32;
        v16h a = load_a_frag(&HLDS[(r1 * 16 + ln) * HID + k0], hi);
        v16h bf = *(const v16h*)&WT1[cj * 128 + k0 + 16 * hi];
        m1 = __builtin_amdgcn_wmma_f32_16x16x32_f16(false, a, false, bf, (short)0, m1, false, false);
    }
    {
        const float bb = mb1[cj];
#pragma unroll
        for (int v = 0; v < 8; ++v)
            M2[(r1 * 16 + hi * 8 + v) * 64 + cj] = (_Float16)fmaxf(m1[v] + bb, 0.0f);
    }
    __syncthreads();

    // ---- MLP layer 2: [32,64] @ [64,64]
    v8f m2 = (v8f)(0.0f);
#pragma unroll
    for (int kt = 0; kt < 2; ++kt) {
        const int k0 = kt * 32;
        v16h a = load_a_frag(&M2[(r1 * 16 + ln) * 64 + k0], hi);
        v16h bf = *(const v16h*)&WT2[cj * 64 + k0 + 16 * hi];
        m2 = __builtin_amdgcn_wmma_f32_16x16x32_f16(false, a, false, bf, (short)0, m2, false, false);
    }
    {
        const float bb = mb2[cj];
#pragma unroll
        for (int v = 0; v < 8; ++v)
            M3[(r1 * 16 + hi * 8 + v) * 64 + cj] = (_Float16)fmaxf(m2[v] + bb, 0.0f);
    }
    __syncthreads();

    // ---- MLP layer 3: [32,64] @ [64,1] -> out
    if (tid < TM) {
        const int n = n0 + tid;
        if (n < N) {
            float s = 0.0f;
#pragma unroll
            for (int k = 0; k < 64; ++k) s += (float)M3[tid * 64 + k] * w3[k];
            out[n] = s + b3[0];
        }
    }
}

// ---------------------------------------------------------------- launch
extern "C" void kernel_launch(void* const* d_in, const int* in_sizes, int n_in,
                              void* d_out, int out_size, void* d_ws, size_t ws_size,
                              hipStream_t stream) {
    const float* x   = (const float*)d_in[0];
    const float* hin = (const float*)d_in[3];
    const float* cin = (const float*)d_in[4];
    const float* Wi  = (const float*)d_in[5],  *Wf = (const float*)d_in[6];
    const float* Wc  = (const float*)d_in[7],  *Wo = (const float*)d_in[8];
    const float* Ci  = (const float*)d_in[9],  *Cf = (const float*)d_in[10];
    const float* Cc  = (const float*)d_in[11], *Co = (const float*)d_in[12];
    const float* cbi = (const float*)d_in[13], *cbf = (const float*)d_in[14];
    const float* cbc = (const float*)d_in[15], *cbo = (const float*)d_in[16];
    const float* wci = (const float*)d_in[17], *wcf = (const float*)d_in[18];
    const float* wco = (const float*)d_in[19];
    const float* bi  = (const float*)d_in[20], *bf = (const float*)d_in[21];
    const float* bc  = (const float*)d_in[22], *bo = (const float*)d_in[23];
    const float* w1  = (const float*)d_in[24], *mb1 = (const float*)d_in[25];
    const float* w2  = (const float*)d_in[26], *mb2 = (const float*)d_in[27];
    const float* w3  = (const float*)d_in[28], *b3  = (const float*)d_in[29];

    const int N = in_sizes[0] / FIN;
    float* out  = (float*)d_out;              // [N]
    float* hout = out + N;                    // [N,128]
    float* cout = hout + (size_t)N * HID;     // [N,128]
    char* ws = (char*)d_ws;

    gclstm_prep<<<(PREP_T + BLOCK - 1) / BLOCK, BLOCK, 0, stream>>>(
        Wi, Wf, Wc, Wo, Ci, Cf, Cc, Co, cbi, cbf, cbc, cbo, bi, bf, bc, bo, w1, w2, ws);

    const int blocks = (N + TM - 1) / TM;     // 15625 for N=500000
    gclstm_main<<<blocks, BLOCK, 0, stream>>>(
        x, hin, cin, wci, wcf, wco, mb1, mb2, w3, b3, ws, out, hout, cout, N);
}
